// Model_50362786513530
// MI455X (gfx1250) — matrix operand order repaired and checked by reading, not compiled, not run on hardware
//
#include <hip/hip_runtime.h>

typedef _Float16 v16h __attribute__((ext_vector_type(16)));
typedef _Float16 h4   __attribute__((ext_vector_type(4)));
typedef float    v8f  __attribute__((ext_vector_type(8)));

#define B_    16
#define S_    512
#define FRAME_ 438
#define DM_   800
#define H_    8
#define DH_   100
#define FFN_  1024
#define HID_  1024
#define POSE_ 274
#define WIN_  100
#define COND_ 10

#define GATE_NT 256   // 4096 / 16 col tiles for gate GEMM
#define GATE_KC 41    // (288 + 1024) / 32 K chunks
#define OUT_NT  18    // ceil(274/16)
#define OUT_KC  57    // 1824 / 32

__device__ __forceinline__ float sigmoidf_(float x) { return 1.0f / (1.0f + __expf(-x)); }

// chunk-local A-fragment layout: identity (contiguous k = 16*half + s, matching B)
__device__ __forceinline__ int fragperm(int k) { return k; }

__device__ __forceinline__ void wave_lds_fence() { asm volatile("s_wait_dscnt 0" ::: "memory"); }

// ---------------------------------------------------------------------------
// fp32-in/out WMMA GEMM: C = epilogue(A(MxK) @ B(KxN)); tile 128x64, BK=32,
// 256 thr = 8 waves; each wave owns a 32x32 output (2x2 WMMA tiles) so each
// K-chunk is 8 ds_load_b128 -> 4 v_wmma (2x the intensity of 1x2 mapping).
// LDS layouts are fragment-major:
//   As: row-major, each 32-chunk permuted  -> af  = one v16h LDS load
//   Bs: column-major (Bs[col][k])          -> bf  = one v16h LDS load
// mode: 0 none, 1 +bias, 2 relu(+bias), 3 +bias+pos_table[(row%S)+1]
// ---------------------------------------------------------------------------
__global__ void wmma_gemm(const float* __restrict__ A, const float* __restrict__ Bm,
                          float* __restrict__ C, const float* __restrict__ bias,
                          const float* __restrict__ pos,
                          int M, int N, int K, int mode)
{
    __shared__ __align__(32) _Float16 As[128 * 32];
    __shared__ __align__(32) _Float16 Bs[64 * 32];   // column-major: Bs[col*32 + k]
    const int tid  = threadIdx.x;
    const int wid  = tid >> 5;
    const int lane = tid & 31;
    const int hi   = lane >> 4;
    const int ln   = lane & 15;
    const int wm   = wid & 3;      // 32-row band
    const int wn   = wid >> 2;     // 32-col band
    const int bm   = blockIdx.y * 128;
    const int bn   = blockIdx.x * 64;

    v8f acc[2][2] = {};
    const int KC = (K + 31) >> 5;
    for (int kc = 0; kc < KC; ++kc) {
        const int k0 = kc << 5;
        const bool kFull = (k0 + 32 <= K);
        // ---- stage A (128x32, permuted chunks)
        if (kFull && bm + 128 <= M) {
            #pragma unroll
            for (int it = 0; it < 4; ++it) {
                int idx = tid + it * 256;                   // 128 rows x 8 float4
                int r = idx >> 3, c4 = (idx & 7) << 2;
                const float4 v = *(const float4*)&A[(size_t)(bm + r) * K + k0 + c4];
                if (k0 + 64 <= K) __builtin_prefetch(&A[(size_t)(bm + r) * K + k0 + 32 + c4], 0, 1);
                h4 hv = { (_Float16)v.x, (_Float16)v.y, (_Float16)v.z, (_Float16)v.w };
                *(h4*)&As[r * 32 + fragperm(c4)] = hv;      // 4-run stays contiguous under perm
            }
        } else {
            for (int i = tid; i < 128 * 32; i += 256) {
                int r = i >> 5, c = i & 31;
                int gr = bm + r, gc = k0 + c;
                float v = (gr < M && gc < K) ? A[(size_t)gr * K + gc] : 0.0f;
                As[r * 32 + fragperm(c)] = (_Float16)v;
            }
        }
        // ---- stage B transposed (Bs[col][k])
        if (kFull && bn + 64 <= N) {
            #pragma unroll
            for (int it = 0; it < 2; ++it) {
                int idx = tid + it * 256;                   // 32 rows x 16 float4
                int r = idx >> 4, c4 = (idx & 15) << 2;
                const float4 v = *(const float4*)&Bm[(size_t)(k0 + r) * N + bn + c4];
                Bs[(c4 + 0) * 32 + r] = (_Float16)v.x;
                Bs[(c4 + 1) * 32 + r] = (_Float16)v.y;
                Bs[(c4 + 2) * 32 + r] = (_Float16)v.z;
                Bs[(c4 + 3) * 32 + r] = (_Float16)v.w;
            }
        } else {
            for (int i = tid; i < 64 * 32; i += 256) {
                int r = i & 31, c = i >> 5;                 // r = k, c = col
                int gr = k0 + r, gc = bn + c;
                float v = (gr < K && gc < N) ? Bm[(size_t)gr * N + gc] : 0.0f;
                Bs[c * 32 + r] = (_Float16)v;
            }
        }
        __syncthreads();
        v16h af[2], bf[2];
        #pragma unroll
        for (int ti = 0; ti < 2; ++ti)
            af[ti] = *(const v16h*)&As[(wm * 32 + ti * 16 + ln) * 32 + 16 * hi];
        #pragma unroll
        for (int tj = 0; tj < 2; ++tj)
            bf[tj] = *(const v16h*)&Bs[(wn * 32 + tj * 16 + ln) * 32 + 16 * hi];
        #pragma unroll
        for (int ti = 0; ti < 2; ++ti)
            #pragma unroll
            for (int tj = 0; tj < 2; ++tj)
                acc[ti][tj] = __builtin_amdgcn_wmma_f32_16x16x32_f16(
                    false, af[ti], false, bf[tj], (short)0, acc[ti][tj], false, false);
        __syncthreads();
    }
    #pragma unroll
    for (int ti = 0; ti < 2; ++ti) {
        #pragma unroll
        for (int r = 0; r < 8; ++r) {
            int m = bm + wm * 32 + ti * 16 + r + 8 * hi;
            if (m >= M) continue;
            int s1 = (m % S_) + 1;
            #pragma unroll
            for (int tj = 0; tj < 2; ++tj) {
                int n = bn + wn * 32 + tj * 16 + ln;
                if (n >= N) continue;
                float v = acc[ti][tj][r];
                if (mode >= 1 && bias) v += bias[n];
                if (mode == 2) v = fmaxf(v, 0.0f);
                if (mode == 3 && pos) v += pos[(size_t)s1 * DM_ + n];
                C[(size_t)m * N + n] = v;
            }
        }
    }
}

// ---------------------------------------------------------------------------
// Banded MHA: one 4-wave workgroup per (b, head, 16-row q tile).
// All fragments are single v16h LDS loads (staged / permuted layouts).
// ---------------------------------------------------------------------------
__global__ void attn_kernel(const float* __restrict__ Q, const float* __restrict__ Km,
                            const float* __restrict__ V, float* __restrict__ O)
{
    __shared__ __align__(32) _Float16 Aq[16 * 128];          // Q tile, permuted chunks
    __shared__ float                  sc[16 * 256];          // scores
    __shared__ __align__(32) _Float16 pr[16 * 256];          // probs, permuted chunks
    __shared__ __align__(32) _Float16 kv[4 * 16 * 256];      // union: K stage / V^T stage
    const int bid  = blockIdx.x;
    const int qt   = bid & 31;
    const int hd   = (bid >> 5) & 7;
    const int b    = bid >> 8;
    const int tid  = threadIdx.x;                            // 128 threads
    const int wid  = tid >> 5;
    const int lane = tid & 31;
    const int hi   = lane >> 4;
    const int ln   = lane & 15;
    const int q0   = qt * 16;
    int lo = q0 - WIN_; if (lo < 0) lo = 0;
    int hiK = q0 + 15 + WIN_; if (hiK > S_ - 1) hiK = S_ - 1;
    const int kt_lo = lo >> 4;
    const int nkt   = (hiK >> 4) - kt_lo + 1;                // <= 15
    const int ncol  = nkt * 16;
    const float scale = 0.1f;                                // 1/sqrt(100)

    // stage Q tile (16 x 128, d>=100 zero), permuted per 32-chunk
    const float* Qb = Q + ((size_t)b * S_ + q0) * DM_ + hd * DH_;
    for (int i = tid; i < 16 * 128; i += 128) {
        int r = i >> 7, d = i & 127;
        float v = (d < DH_) ? Qb[(size_t)r * DM_ + d] : 0.0f;
        Aq[r * 128 + (d & ~31) + fragperm(d & 31)] = (_Float16)v;
    }
    for (int i = tid; i < 16 * 256; i += 128) pr[i] = (_Float16)0.0f;
    __syncthreads();
    v16h aq[4];
    #pragma unroll
    for (int c = 0; c < 4; ++c)
        aq[c] = *(const v16h*)&Aq[ln * 128 + c * 32 + 16 * hi];

    // phase 1: banded score tiles (Q @ K^T)
    _Float16* Kst = kv + wid * 2048;                         // per-wave 16x128
    for (int it = wid; it < nkt; it += 4) {
        int kt = kt_lo + it;
        const float* Kb = Km + ((size_t)b * S_ + kt * 16) * DM_ + hd * DH_;
        for (int i = lane; i < 16 * 128; i += 32) {
            int r = i >> 7, d = i & 127;
            float v = (d < DH_) ? Kb[(size_t)r * DM_ + d] : 0.0f;
            Kst[i] = (_Float16)v;
        }
        wave_lds_fence();
        v8f acc = {};
        #pragma unroll
        for (int c = 0; c < 4; ++c) {
            const v16h bf = *(const v16h*)&Kst[ln * 128 + c * 32 + 16 * hi];
            acc = __builtin_amdgcn_wmma_f32_16x16x32_f16(false, aq[c], false, bf, (short)0, acc, false, false);
        }
        #pragma unroll
        for (int r = 0; r < 8; ++r) {
            int m = r + 8 * hi;
            int d = (q0 + m) - (kt * 16 + ln); if (d < 0) d = -d;
            float s = acc[r] * scale;
            if (d > WIN_) s = -1e9f;
            sc[m * 256 + it * 16 + ln] = s;
        }
        wave_lds_fence();
    }
    __syncthreads();
    // phase 2: row softmax; probs written permuted per 32-chunk
    if (tid < 16) {
        int m = tid;
        float mx = -1e30f;
        for (int j = 0; j < ncol; ++j) mx = fmaxf(mx, sc[m * 256 + j]);
        float sum = 0.0f;
        for (int j = 0; j < ncol; ++j) { float e = __expf(sc[m * 256 + j] - mx); sc[m * 256 + j] = e; sum += e; }
        float inv = 1.0f / sum;
        for (int j = 0; j < ncol; ++j)
            pr[m * 256 + (j & ~31) + fragperm(j & 31)] = (_Float16)(sc[m * 256 + j] * inv);
    }
    __syncthreads();
    // phase 3: O = P @ V  (7 d-tiles of 16; per-wave V^T stage)
    const int nchunks = (ncol + 31) >> 5;
    _Float16* Vt = kv + wid * 4096;                          // per-wave 16 x 256 (d x key)
    for (int nt = wid; nt < 7; nt += 4) {
        for (int i = lane; i < 16 * 256; i += 32) {
            int key = i >> 4, dl = i & 15;
            int d = nt * 16 + dl;
            float v = (key < ncol && d < DH_)
                        ? V[((size_t)b * S_ + kt_lo * 16 + key) * DM_ + hd * DH_ + d] : 0.0f;
            Vt[dl * 256 + key] = (_Float16)v;
        }
        wave_lds_fence();
        v8f acc = {};
        for (int c = 0; c < nchunks; ++c) {
            const v16h af = *(const v16h*)&pr[ln * 256 + c * 32 + 16 * hi];
            const v16h bf = *(const v16h*)&Vt[ln * 256 + c * 32 + 16 * hi];
            acc = __builtin_amdgcn_wmma_f32_16x16x32_f16(false, af, false, bf, (short)0, acc, false, false);
        }
        #pragma unroll
        for (int r = 0; r < 8; ++r) {
            int m = r + 8 * hi, d = nt * 16 + ln;
            if (d < DH_) O[((size_t)b * S_ + q0 + m) * DM_ + hd * DH_ + d] = acc[r];
        }
        wave_lds_fence();
    }
}

// ---------------------------------------------------------------------------
__global__ void ln_residual(float* __restrict__ X, const float* __restrict__ T,
                            const float* __restrict__ g, const float* __restrict__ bt)
{
    __shared__ float buf[DM_];
    __shared__ float red[256];
    const size_t row = blockIdx.x;
    const int tid = threadIdx.x;
    float lsum = 0.0f;
    for (int i = tid; i < DM_; i += 256) {
        float v = X[row * DM_ + i] + T[row * DM_ + i];
        buf[i] = v; lsum += v;
    }
    red[tid] = lsum; __syncthreads();
    for (int s = 128; s > 0; s >>= 1) { if (tid < s) red[tid] += red[tid + s]; __syncthreads(); }
    float mean = red[0] / (float)DM_;
    __syncthreads();
    float lvar = 0.0f;
    for (int i = tid; i < DM_; i += 256) { float d = buf[i] - mean; lvar += d * d; }
    red[tid] = lvar; __syncthreads();
    for (int s = 128; s > 0; s >>= 1) { if (tid < s) red[tid] += red[tid + s]; __syncthreads(); }
    float inv = rsqrtf(red[0] / (float)DM_ + 1e-6f);
    for (int i = tid; i < DM_; i += 256)
        X[row * DM_ + i] = (buf[i] - mean) * inv * g[i] + bt[i];
}

// ---------------------------------------------------------------------------
// One-time weight swizzles into this file's B-fragment order (contiguous k, matching A):
// out[((nt*KC + c)*32 + lane)*16 + e] = W[k = c*32 + e + 16*(lane>>4)][n = nt*16 + (lane&15)]
// ---------------------------------------------------------------------------
__global__ void swizzle_gate_w(const float* __restrict__ Wx, const float* __restrict__ Wh,
                               _Float16* __restrict__ out)
{
    size_t idx = (size_t)blockIdx.x * 256 + threadIdx.x;
    if (idx >= (size_t)GATE_NT * GATE_KC * 512) return;
    int e    = idx & 15;
    int lane = (idx >> 4) & 31;
    int c    = (int)((idx >> 9) % GATE_KC);
    int nt   = (int)(idx / ((size_t)GATE_KC * 512));
    int n    = nt * 16 + (lane & 15);
    int kk   = c * 32 + e + 16 * (lane >> 4);
    float v = 0.0f;
    if (kk < POSE_)      v = Wx[(size_t)kk * 4096 + n];
    else if (kk >= 288)  v = Wh[(size_t)(kk - 288) * 4096 + n];
    out[idx] = (_Float16)v;
}

__global__ void swizzle_out_w(const float* __restrict__ W, _Float16* __restrict__ out)
{
    size_t idx = (size_t)blockIdx.x * 256 + threadIdx.x;
    if (idx >= (size_t)OUT_NT * OUT_KC * 512) return;
    int e    = idx & 15;
    int lane = (idx >> 4) & 31;
    int c    = (int)((idx >> 9) % OUT_KC);
    int nt   = (int)(idx / ((size_t)OUT_KC * 512));
    int n    = nt * 16 + (lane & 15);
    int kk   = c * 32 + e + 16 * (lane >> 4);           // < 1824 always
    float v = (n < POSE_) ? W[(size_t)kk * POSE_ + n] : 0.0f;
    out[idx] = (_Float16)v;
}

// ---------------------------------------------------------------------------
// Persistent single-workgroup LSTM decoder (512 thr = 16 waves, M=16=one tile).
// gates: K = 1312 (41 chunks), N = 4096 (256 tiles); out: K = 1824 (57), N = 274 (18)
// A-fragments: permuted LDS, one v16h load, hoisted over groups of 4 N-tiles.
// B-fragments: swizzled global, one v16h load (wave loads 1KB contiguous).
// ---------------------------------------------------------------------------
__global__ void lstm_kernel(const float* __restrict__ tgt, const float* __restrict__ enc,
                            const float* __restrict__ h0, const float* __restrict__ c0,
                            const float* __restrict__ dec0,
                            const _Float16* __restrict__ Wg, const _Float16* __restrict__ Wo,
                            const float* __restrict__ lb, const float* __restrict__ ob,
                            const int* __restrict__ epoch_p,
                            float* __restrict__ gates, float* __restrict__ hws,
                            float* __restrict__ cws, float* __restrict__ prev,
                            float* __restrict__ out)
{
    __shared__ __align__(32) _Float16 sbuf[16 * 1824];  // union: xh (16x1312) / he (16x1824)
    const int tid  = threadIdx.x;                       // 0..511
    const int wid  = tid >> 5;
    const int lane = tid & 31;
    const int hi   = lane >> 4;
    const int ln   = lane & 15;
    const int p    = epoch_p[0] / 100;                  // int(epoch_i * LAMB)
    const int period = p + COND_;
    const v16h* Wgv = (const v16h*)Wg;
    const v16h* Wov = (const v16h*)Wo;

    for (int i = tid; i < B_ * HID_; i += 512) { hws[i] = h0[i]; cws[i] = c0[i]; }
    for (int i = tid; i < B_ * POSE_; i += 512) prev[i] = dec0[i];
    __threadfence_block();
    __syncthreads();

    for (int t = 0; t < S_; ++t) {
        const bool msk = (t % period) >= p;
        // ---- phase 1: xh = [x_t | pad | h] f16, chunk-permuted
        for (int i = tid; i < B_ * 288; i += 512) {
            int bb = i / 288, c = i % 288;
            float v = 0.0f;
            if (c < POSE_)
                v = msk ? tgt[((size_t)bb * S_ + t) * POSE_ + c] : prev[bb * POSE_ + c];
            sbuf[bb * 1312 + (c & ~31) + fragperm(c & 31)] = (_Float16)v;
        }
        for (int i = tid; i < B_ * HID_; i += 512) {
            int bb = i / HID_, c = i % HID_;
            sbuf[bb * 1312 + 288 + (c & ~31) + fragperm(c & 31)] = (_Float16)hws[i];
        }
        __syncthreads();
        // ---- phase 2: gates = xh @ [Wx;Wh] + b  (4 N-tiles per wave-group, af hoisted)
        for (int g = 0; g < 4; ++g) {
            const int nt0 = wid + 64 * g;               // tiles nt0 + 16*u, u=0..3
            v8f acc[4] = {};
            for (int c = 0; c < GATE_KC; ++c) {
                const v16h af = *(const v16h*)&sbuf[ln * 1312 + c * 32 + 16 * hi];
                #pragma unroll
                for (int u = 0; u < 4; ++u) {
                    const int nt = nt0 + 16 * u;
                    const v16h bf = Wgv[((size_t)nt * GATE_KC + c) * 32 + lane];
                    acc[u] = __builtin_amdgcn_wmma_f32_16x16x32_f16(
                        false, af, false, bf, (short)0, acc[u], false, false);
                }
            }
            #pragma unroll
            for (int u = 0; u < 4; ++u) {
                int n = (nt0 + 16 * u) * 16 + ln;
                float bv = lb[n];
                #pragma unroll
                for (int r = 0; r < 8; ++r)
                    gates[(r + 8 * hi) * 4096 + n] = acc[u][r] + bv;
            }
        }
        __threadfence_block();
        __syncthreads();
        // ---- phase 3: LSTM cell + stage [h | enc_t] f16, chunk-permuted
        for (int i = tid; i < B_ * HID_; i += 512) {
            int bb = i / HID_, jj = i % HID_;
            float ig = gates[bb * 4096 + jj];
            float fg = gates[bb * 4096 + 1024 + jj];
            float gg = gates[bb * 4096 + 2048 + jj];
            float og = gates[bb * 4096 + 3072 + jj];
            float cc = sigmoidf_(fg) * cws[i] + sigmoidf_(ig) * tanhf(gg);
            float hh = sigmoidf_(og) * tanhf(cc);
            cws[i] = cc; hws[i] = hh;
            sbuf[bb * 1824 + (jj & ~31) + fragperm(jj & 31)] = (_Float16)hh;
        }
        for (int i = tid; i < B_ * DM_; i += 512) {
            int bb = i / DM_, dd = i % DM_;
            sbuf[bb * 1824 + 1024 + (dd & ~31) + fragperm(dd & 31)] =
                (_Float16)enc[((size_t)bb * S_ + t) * DM_ + dd];
        }
        __syncthreads();
        // ---- phase 4: out = [h|enc] @ Wout + ob
        for (int nt = wid; nt < OUT_NT; nt += 16) {
            int n = nt * 16 + ln;
            v8f acc = {};
            for (int c = 0; c < OUT_KC; ++c) {
                const v16h af = *(const v16h*)&sbuf[ln * 1824 + c * 32 + 16 * hi];
                const v16h bf = Wov[((size_t)nt * OUT_KC + c) * 32 + lane];
                acc = __builtin_amdgcn_wmma_f32_16x16x32_f16(false, af, false, bf, (short)0, acc, false, false);
            }
            if (n < POSE_) {
                float bv = ob[n];
                #pragma unroll
                for (int r = 0; r < 8; ++r) {
                    int bb = r + 8 * hi;
                    float v = acc[r] + bv;
                    out[((size_t)bb * S_ + t) * POSE_ + n] = v;
                    prev[bb * POSE_ + n] = v;
                }
            }
        }
        __threadfence_block();
        __syncthreads();
    }
}

// ---------------------------------------------------------------------------
extern "C" void kernel_launch(void* const* d_in, const int* in_sizes, int n_in,
                              void* d_out, int out_size, void* d_ws, size_t ws_size,
                              hipStream_t stream)
{
    (void)in_sizes; (void)n_in; (void)out_size; (void)ws_size;
    const float* src  = (const float*)d_in[0];
    const float* tgt  = (const float*)d_in[2];
    const float* h0   = (const float*)d_in[3];
    const float* c0   = (const float*)d_in[4];
    const float* dec0 = (const float*)d_in[5];
    const float* embW = (const float*)d_in[6];
    const float* embB = (const float*)d_in[7];
    const float* posT = (const float*)d_in[8];
    const float* Wq   = (const float*)d_in[9];
    const float* Wk   = (const float*)d_in[10];
    const float* Wv   = (const float*)d_in[11];
    const float* Wo   = (const float*)d_in[12];
    const float* ln1g = (const float*)d_in[13];
    const float* ln1b = (const float*)d_in[14];
    const float* fW1  = (const float*)d_in[15];
    const float* fb1  = (const float*)d_in[16];
    const float* fW2  = (const float*)d_in[17];
    const float* fb2  = (const float*)d_in[18];
    const float* ln2g = (const float*)d_in[19];
    const float* ln2b = (const float*)d_in[20];
    const float* lWx  = (const float*)d_in[21];
    const float* lWh  = (const float*)d_in[22];
    const float* lb   = (const float*)d_in[23];
    const float* oW   = (const float*)d_in[24];
    const float* ob   = (const float*)d_in[25];
    const int*   ep   = (const int*)d_in[26];

    char* ws = (char*)d_ws;
    size_t off = 0;
    auto alloc = [&](size_t bytes) -> void* {
        void* ptr = ws + off;
        off = (off + bytes + 255) & ~(size_t)255;
        return ptr;
    };
    const size_t MR = (size_t)B_ * S_;
    float* X   = (float*)alloc(MR * DM_ * 4);
    float* Qb  = (float*)alloc(MR * DM_ * 4);
    float* Kb  = (float*)alloc(MR * DM_ * 4);
    float* Vb  = (float*)alloc(MR * DM_ * 4);
    float* T1  = (float*)alloc(MR * FFN_ * 4);
    _Float16* Wg16 = (_Float16*)alloc((size_t)GATE_NT * GATE_KC * 512 * 2);
    _Float16* Wo16 = (_Float16*)alloc((size_t)OUT_NT * OUT_KC * 512 * 2);
    float* gates = (float*)alloc((size_t)B_ * 4 * HID_ * 4);
    float* hws   = (float*)alloc((size_t)B_ * HID_ * 4);
    float* cws   = (float*)alloc((size_t)B_ * HID_ * 4);
    float* prev  = (float*)alloc((size_t)B_ * POSE_ * 4);

    const dim3 blk(256);
    const dim3 gD(13, 64);    // N=800 tiles x M=8192/128
    const dim3 gF(16, 64);    // N=1024

    wmma_gemm<<<gD, blk, 0, stream>>>(src, embW, X, embB, posT, (int)MR, DM_, FRAME_, 3);
    wmma_gemm<<<gD, blk, 0, stream>>>(X, Wq, Qb, nullptr, nullptr, (int)MR, DM_, DM_, 0);
    wmma_gemm<<<gD, blk, 0, stream>>>(X, Wk, Kb, nullptr, nullptr, (int)MR, DM_, DM_, 0);
    wmma_gemm<<<gD, blk, 0, stream>>>(X, Wv, Vb, nullptr, nullptr, (int)MR, DM_, DM_, 0);
    attn_kernel<<<dim3(B_ * H_ * (S_ / 16)), dim3(128), 0, stream>>>(Qb, Kb, Vb, T1);
    wmma_gemm<<<gD, blk, 0, stream>>>(T1, Wo, Qb, nullptr, nullptr, (int)MR, DM_, DM_, 0);
    ln_residual<<<dim3((unsigned)MR), blk, 0, stream>>>(X, Qb, ln1g, ln1b);
    wmma_gemm<<<gF, blk, 0, stream>>>(X, fW1, T1, fb1, nullptr, (int)MR, FFN_, DM_, 2);
    wmma_gemm<<<gD, blk, 0, stream>>>(T1, fW2, Qb, fb2, nullptr, (int)MR, DM_, FFN_, 1);
    ln_residual<<<dim3((unsigned)MR), blk, 0, stream>>>(X, Qb, ln2g, ln2b);

    const size_t nG = (size_t)GATE_NT * GATE_KC * 512;
    const size_t nO = (size_t)OUT_NT * OUT_KC * 512;
    swizzle_gate_w<<<dim3((unsigned)((nG + 255) / 256)), blk, 0, stream>>>(lWx, lWh, Wg16);
    swizzle_out_w<<<dim3((unsigned)((nO + 255) / 256)), blk, 0, stream>>>(oW, Wo16);

    lstm_kernel<<<dim3(1), dim3(512), 0, stream>>>(tgt, X, h0, c0, dec0, Wg16, Wo16,
                                                   lb, ob, ep, gates, hws, cws, prev,
                                                   (float*)d_out);
}
